// VectorField_g_86973087744760
// MI455X (gfx1250) — matrix operand order repaired and checked by reading, not compiled, not run on hardware
//
#include <hip/hip_runtime.h>
#include <math.h>

// ---------------------------------------------------------------------------
// Sizes (compile-time, match the reference)
// ---------------------------------------------------------------------------
#define BB 64      // batch
#define NN 2048    // nodes
#define CC 64      // hidden (in/out channels)
#define HH 64      // hidden_hidden
#define DD 16      // embed dim
#define KK 3       // cheb_k
#define BH (BB*HH) // 4096

typedef __bf16 bf16_t;
typedef __attribute__((ext_vector_type(16))) __bf16 v16bf;
typedef __attribute__((ext_vector_type(8)))  __bf16 v8bf;
typedef __attribute__((ext_vector_type(8)))  float  v8f;
typedef __attribute__((ext_vector_type(4)))  int    v4i;

#define GLOBAL_AS __attribute__((address_space(1)))
#define LDS_AS    __attribute__((address_space(3)))

// gfx1250 async global->LDS path (guarded; falls back to direct loads)
#if defined(__gfx1250__) && __has_builtin(__builtin_amdgcn_global_load_async_to_lds_b128)
#define USE_ASYNC_LDS 1
#else
#define USE_ASYNC_LDS 0
#endif

// ---------------------------------------------------------------------------
// Helpers
// ---------------------------------------------------------------------------
__device__ __forceinline__ v16bf cat8(v8bf a, v8bf b) {
    return __builtin_shufflevector(a, b, 0,1,2,3,4,5,6,7,8,9,10,11,12,13,14,15);
}
__device__ __forceinline__ v8bf cvt8(const float* __restrict__ p) {
    v8bf r;
#pragma unroll
    for (int i = 0; i < 8; ++i) r[i] = (bf16_t)p[i];
    return r;
}
__device__ __forceinline__ v16bf ld16bf(const bf16_t* p) {
    return cat8(*(const v8bf*)p, *(const v8bf*)(p + 8));
}
__device__ __forceinline__ v8f wmma_bf16(v16bf a, v16bf b, v8f c) {
    // (neg_a, A, neg_b, B, c_mod, C, reuse_a, reuse_b)
    return __builtin_amdgcn_wmma_f32_16x16x32_bf16(false, a, false, b, (short)0, c,
                                                   false, false);
}
__device__ __forceinline__ v8f zero8() {
    v8f z;
#pragma unroll
    for (int i = 0; i < 8; ++i) z[i] = 0.f;
    return z;
}

#if USE_ASYNC_LDS
__device__ __forceinline__ void async_cp16(bf16_t* lds, const bf16_t* g) {
    // prototype: (v4i as(1)*, v4i as(3)*, imm offset, imm cpol)
    __builtin_amdgcn_global_load_async_to_lds_b128(
        (GLOBAL_AS v4i*)g, (LDS_AS v4i*)lds, 0, 0);
}
__device__ __forceinline__ void wait_async0() {
#if __has_builtin(__builtin_amdgcn_s_wait_asynccnt)
    __builtin_amdgcn_s_wait_asynccnt(0);
#else
    asm volatile("s_wait_asynccnt 0x0" ::: "memory");
#endif
}
#endif

// ---------------------------------------------------------------------------
// Kernel 0: one-time f32 -> bf16 conversion of W_in / W_out (4096 elems each)
// ---------------------------------------------------------------------------
__global__ __launch_bounds__(256) void prep_weights_kernel(
    const float* __restrict__ W_in, const float* __restrict__ W_out,
    bf16_t* __restrict__ Wb)
{
    const int t = blockIdx.x*256 + threadIdx.x;      // 8192 total
    if (t < HH*CC) Wb[t] = (bf16_t)W_in[t];
    else           Wb[t] = (bf16_t)W_out[t - HH*CC];
}

// ---------------------------------------------------------------------------
// Kernel 1: A = softmax(relu(E1 E2^T - E2 E1^T), axis=1)  -> bf16 [N,N]
// ---------------------------------------------------------------------------
__global__ __launch_bounds__(256) void adj_softmax_kernel(
    const float* __restrict__ E1, const float* __restrict__ E2,
    bf16_t* __restrict__ A)
{
    __shared__ float red[256];
    __shared__ float e1s[DD], e2s[DD];
    const int n = blockIdx.x, tid = threadIdx.x;
    if (tid < DD) { e1s[tid] = E1[n*DD + tid]; e2s[tid] = E2[n*DD + tid]; }
    __syncthreads();

    float l[8];
    float mx = -1e30f;
#pragma unroll
    for (int j = 0; j < 8; ++j) {
        const int m = tid + j*256;
        const float* r1 = E2 + (size_t)m*DD;
        const float* r2 = E1 + (size_t)m*DD;
        float d1 = 0.f, d2 = 0.f;
#pragma unroll
        for (int d = 0; d < DD; ++d) { d1 += e1s[d]*r1[d]; d2 += e2s[d]*r2[d]; }
        float v = fmaxf(d1 - d2, 0.f);
        l[j] = v; mx = fmaxf(mx, v);
    }
    red[tid] = mx; __syncthreads();
    for (int s = 128; s > 0; s >>= 1) {
        if (tid < s) red[tid] = fmaxf(red[tid], red[tid+s]);
        __syncthreads();
    }
    mx = red[0]; __syncthreads();

    float e[8], sum = 0.f;
#pragma unroll
    for (int j = 0; j < 8; ++j) { e[j] = __expf(l[j] - mx); sum += e[j]; }
    red[tid] = sum; __syncthreads();
    for (int s = 128; s > 0; s >>= 1) {
        if (tid < s) red[tid] += red[tid+s];
        __syncthreads();
    }
    const float inv = 1.f / red[0];
#pragma unroll
    for (int j = 0; j < 8; ++j)
        A[(size_t)n*NN + tid + j*256] = (bf16_t)(e[j]*inv);
}

// ---------------------------------------------------------------------------
// Kernel 2: h = relu(z @ W_in^T + b_in), stored transposed as hT[m, b*H+hh]
// ---------------------------------------------------------------------------
__global__ __launch_bounds__(256) void linear_in_kernel(
    const float* __restrict__ z, const bf16_t* __restrict__ Wb_in,
    const float* __restrict__ b_in, bf16_t* __restrict__ hT)
{
    const int tid = threadIdx.x;
    const int wave = tid >> 5, lane = tid & 31;
    const int l16 = lane & 15, half = lane >> 4;
    const int rowA = blockIdx.x*128 + wave*16 + l16;   // A-matrix row (M = l16)

    v8f acc[4] = { zero8(), zero8(), zero8(), zero8() };

#pragma unroll
    for (int kc = 0; kc < 2; ++kc) {                   // K chunks of 32
        const int c0 = kc*32;
        const float* zr = z + (size_t)rowA*CC + c0;
        v16bf av = cat8(cvt8(zr + 16*half), cvt8(zr + 16*half + 8));
#pragma unroll
        for (int tc = 0; tc < 4; ++tc) {
            const int hh = tc*16 + l16;                // B column (fixed / lane)
            // B(k=c, n=hh) = W_in[hh, c]: contiguous bf16 row read
            v16bf bv = ld16bf(Wb_in + (size_t)hh*CC + c0 + 16*half);
            acc[tc] = wmma_bf16(av, bv, acc[tc]);
        }
    }
#pragma unroll
    for (int tc = 0; tc < 4; ++tc) {
        const int hh = tc*16 + l16;
        const float bias = b_in[hh];
#pragma unroll
        for (int v = 0; v < 8; ++v) {
            const int r = blockIdx.x*128 + wave*16 + v + 8*half;  // D row M=v+8*half
            const float val = fmaxf(acc[tc][v] + bias, 0.f);
            const int b = r >> 11, m = r & (NN-1);
            hT[(size_t)m*BH + b*HH + hh] = (bf16_t)val;
        }
    }
}

// ---------------------------------------------------------------------------
// Kernel 3: generic bf16 GEMM  C = op(A @ B)   (f32 accumulate)
//   64x64 tile/WG; A tile async-staged to LDS; B tile staged transposed;
//   next K-tile prefetched.  epi==1: C = 2*(A@B) - I.
// ---------------------------------------------------------------------------
__global__ __launch_bounds__(256) void gemm_bf16_kernel(
    const bf16_t* __restrict__ A, const bf16_t* __restrict__ Bm,
    bf16_t* __restrict__ Cm, int Kd, int lda, int ldb, int ldc, int epi)
{
    __shared__ __align__(16) bf16_t BT[64*40];   // [col][kd], stride 40 (80B)
#if USE_ASYNC_LDS
    __shared__ __align__(16) bf16_t As[64*32];   // [row][kd]
#endif
    const int tid = threadIdx.x;
    const int wave = tid >> 5, lane = tid & 31;
    const int l16 = lane & 15, half = lane >> 4;
    const int rowBase = blockIdx.y*64, colBase = blockIdx.x*64;
    const int tr = wave >> 1, tcb = (wave & 1)*2;
    const int kl = (tid*8) >> 6;                 // B-stage: local kd row (0..31)
    const int c0 = (tid*8) & 63;                 // B-stage: 8 consecutive cols
    const int arow = tid >> 2, ach = (tid & 3)*8; // A-stage: row, 8-elem chunk

    v8f acc[2] = { zero8(), zero8() };

    for (int kd0 = 0; kd0 < Kd; kd0 += 32) {
#if USE_ASYNC_LDS
        // async DMA A[rowBase..+64, kd0..+32] into LDS (1 x b128 per thread)
        async_cp16(&As[arow*32 + ach],
                   A + (size_t)(rowBase + arow)*lda + kd0 + ach);
#endif
        // stage B[kd0..+32, colBase..+64] transposed into LDS
        {
            v8bf v = *(const v8bf*)(Bm + (size_t)(kd0+kl)*ldb + colBase + c0);
#pragma unroll
            for (int i = 0; i < 8; ++i) BT[(c0+i)*40 + kl] = v[i];
        }
        if (kd0 + 32 < Kd) {   // prefetch next K-tile (global_prefetch_b8)
            __builtin_prefetch((const void*)(Bm + (size_t)(kd0+32+kl)*ldb + colBase + c0), 0, 3);
            __builtin_prefetch((const void*)(A + (size_t)(rowBase + arow)*lda + kd0 + 32 + ach), 0, 3);
        }
#if USE_ASYNC_LDS
        wait_async0();
#endif
        __syncthreads();

        // A fragment: one contiguous 16-elem run, k = 16*half + s
#if USE_ASYNC_LDS
        const bf16_t* ar = &As[(tr*16 + l16)*32];
        v16bf av = cat8(*(const v8bf*)(ar + 16*half),
                        *(const v8bf*)(ar + 16*half + 8));
#else
        const bf16_t* ar = A + (size_t)(rowBase + tr*16 + l16)*lda + kd0;
        v16bf av = cat8(*(const v8bf*)(ar + 16*half),
                        *(const v8bf*)(ar + 16*half + 8));
#endif
#pragma unroll
        for (int j = 0; j < 2; ++j) {
            const int col = (tcb+j)*16 + l16;
            const bf16_t* bp = &BT[col*40 + 16*half];
            v16bf bv = cat8(*(const v8bf*)bp, *(const v8bf*)(bp + 8));
            acc[j] = wmma_bf16(av, bv, acc[j]);
        }
        __syncthreads();
    }

#pragma unroll
    for (int j = 0; j < 2; ++j) {
        const int col = colBase + (tcb+j)*16 + l16;
#pragma unroll
        for (int v = 0; v < 8; ++v) {
            const int row = rowBase + tr*16 + v + 8*half;
            float val = acc[j][v];
            if (epi) val = 2.f*val - ((row == col) ? 1.f : 0.f);
            Cm[(size_t)row*ldc + col] = (bf16_t)val;
        }
    }
}

// ---------------------------------------------------------------------------
// Kernel 4: per-node fused tail. One WG (256 thr) per node n:
//   X strip (hT/Y1/Y2 rows of node n, 3x4096 bf16) async-staged to LDS while
//   the per-node weights WnT are built; then two chained WMMA GEMMs:
//   [64x192]x[192x64] -> tanh -> [64x64]x[64x64] -> tanh -> out.
// ---------------------------------------------------------------------------
__global__ __launch_bounds__(256) void node_kernel(
    const bf16_t* __restrict__ hT, const bf16_t* __restrict__ Y1,
    const bf16_t* __restrict__ Y2,
    const float* __restrict__ E1, const float* __restrict__ E2,
    const float* __restrict__ wp1, const float* __restrict__ wp2,
    const float* __restrict__ bp1, const float* __restrict__ bp2,
    const bf16_t* __restrict__ Wb_out, const float* __restrict__ b_out,
    float* __restrict__ out)
{
    __shared__ __align__(16) bf16_t WnT[HH * (KK*HH)];  // [o][ki] 64x192, 24 KB
    __shared__ __align__(16) bf16_t h2s[HH * HH];       // [b][o]  64x64,   8 KB
#if USE_ASYNC_LDS
    __shared__ __align__(16) bf16_t Xs[KK * BH];        // [k][b*64+i],    24 KB
#endif
    __shared__ float biasS[HH];
    __shared__ float e1s[DD], e2s[DD];

    const int n = blockIdx.x, tid = threadIdx.x;

#if USE_ASYNC_LDS
    // kick off DMA of the node's Chebyshev feature strip (overlaps W-combine)
#pragma unroll
    for (int c = 0; c < 6; ++c) {                 // 1536 x b128 chunks / 256 thr
        const int e  = (c*256 + tid);
        const int k  = e >> 9;                    // 512 chunks per k
        const int off = (e & 511)*8;
        const bf16_t* src = ((k == 0) ? hT : (k == 1) ? Y1 : Y2) + (size_t)n*BH + off;
        async_cp16(&Xs[k*BH + off], src);
    }
#endif

    if (tid < DD) { e1s[tid] = E1[n*DD + tid]; e2s[tid] = E2[n*DD + tid]; }
    __syncthreads();

    // ---- build per-node weights (transposed) and bias in LDS ----
    for (int idx = tid; idx < HH*KK*HH; idx += 256) {
        const int o = idx / (KK*HH), ki = idx % (KK*HH);
        const int k = ki >> 6, i = ki & 63;
        float a = 0.f;
#pragma unroll
        for (int d = 0; d < DD; ++d) {
            const size_t off = (size_t)(d*KK + k)*4096 + (size_t)i*64 + o;
            a += e1s[d]*wp1[off] - e2s[d]*wp2[off];
        }
        WnT[o*(KK*HH) + ki] = (bf16_t)a;
    }
    if (tid < HH) {
        float a = 0.f;
#pragma unroll
        for (int d = 0; d < DD; ++d)
            a += e1s[d]*bp1[d*HH + tid] - e2s[d]*bp2[d*HH + tid];
        biasS[tid] = a;
    }
#if USE_ASYNC_LDS
    wait_async0();
#endif
    __syncthreads();

    const int wave = tid >> 5, lane = tid & 31;
    const int l16 = lane & 15, half = lane >> 4;
    const int tr = wave >> 1, tcb = (wave & 1)*2;

    // ---- GEMM1: [B=64 x 192] x [192 x 64] -> h2 ----
    v8f acc[2] = { zero8(), zero8() };
#pragma unroll
    for (int kc = 0; kc < 6; ++kc) {                 // 192 / 32
        const int k  = kc >> 1;
        const int i0 = (kc & 1)*32;
#if USE_ASYNC_LDS
        const bf16_t* ar = &Xs[k*BH + (tr*16 + l16)*HH + i0];
#else
        const bf16_t* Yk = (k == 0) ? hT : ((k == 1) ? Y1 : Y2);
        const bf16_t* ar = Yk + (size_t)n*BH + (size_t)(tr*16 + l16)*HH + i0;
#endif
        v16bf av = cat8(*(const v8bf*)(ar + 16*half),
                        *(const v8bf*)(ar + 16*half + 8));
#pragma unroll
        for (int j = 0; j < 2; ++j) {
            const int o = (tcb+j)*16 + l16;
            const bf16_t* bp = &WnT[o*(KK*HH) + kc*32 + 16*half];
            v16bf bv = cat8(*(const v8bf*)bp, *(const v8bf*)(bp + 8));
            acc[j] = wmma_bf16(av, bv, acc[j]);
        }
    }
#pragma unroll
    for (int j = 0; j < 2; ++j) {
        const int o = (tcb+j)*16 + l16;
        const float bs = biasS[o];
#pragma unroll
        for (int v = 0; v < 8; ++v) {
            const int b = tr*16 + v + 8*half;
            h2s[b*HH + o] = (bf16_t)tanhf(acc[j][v] + bs);
        }
    }
    __syncthreads();

    // ---- GEMM2: h2 [64x64] x W_out^T [64x64] -> out ----
    v8f acc2[2] = { zero8(), zero8() };
#pragma unroll
    for (int kc = 0; kc < 2; ++kc) {
        const int o0 = kc*32;
        const bf16_t* ar = &h2s[(tr*16 + l16)*HH + o0];
        v16bf av = cat8(*(const v8bf*)(ar + 16*half),
                        *(const v8bf*)(ar + 16*half + 8));
#pragma unroll
        for (int j = 0; j < 2; ++j) {
            const int c = (tcb+j)*16 + l16;
            // B(k=o, n=c) = W_out[c, o]: contiguous bf16 row read
            v16bf bv = ld16bf(Wb_out + (size_t)c*HH + o0 + 16*half);
            acc2[j] = wmma_bf16(av, bv, acc2[j]);
        }
    }
#pragma unroll
    for (int j = 0; j < 2; ++j) {
        const int c = (tcb+j)*16 + l16;
        const float bo = b_out[c];
#pragma unroll
        for (int v = 0; v < 8; ++v) {
            const int b = tr*16 + v + 8*half;
            out[((size_t)b*NN + n)*CC + c] = tanhf(acc2[j][v] + bo);
        }
    }
}

// ---------------------------------------------------------------------------
// Launch: workspace layout (needs 64 MB + 16 KB):
//   [ 0,  8M) A_bf   bf16 [N,N]
//   [ 8M,16M) T2_bf  bf16 [N,N]      = 2A^2 - I
//   [16M,32M) hT     bf16 [N, B*H]
//   [32M,48M) Y1     bf16 [N, B*H]   = A @ hT
//   [48M,64M) Y2     bf16 [N, B*H]   = T2 @ hT
//   [64M, +16K) Wb   bf16 W_in(4096) ++ W_out(4096)
// ---------------------------------------------------------------------------
extern "C" void kernel_launch(void* const* d_in, const int* in_sizes, int n_in,
                              void* d_out, int out_size, void* d_ws, size_t ws_size,
                              hipStream_t stream) {
    const float* z     = (const float*)d_in[1];
    const float* E1    = (const float*)d_in[2];
    const float* E2    = (const float*)d_in[3];
    const float* W_in  = (const float*)d_in[4];
    const float* b_in  = (const float*)d_in[5];
    const float* W_out = (const float*)d_in[6];
    const float* b_out = (const float*)d_in[7];
    const float* wp1   = (const float*)d_in[8];
    const float* wp2   = (const float*)d_in[9];
    const float* bp1   = (const float*)d_in[10];
    const float* bp2   = (const float*)d_in[11];
    float* out = (float*)d_out;

    char* ws = (char*)d_ws;
    bf16_t* A_bf   = (bf16_t*)(ws);
    bf16_t* T2_bf  = (bf16_t*)(ws + (size_t)( 8u<<20));
    bf16_t* hT     = (bf16_t*)(ws + (size_t)(16u<<20));
    bf16_t* Y1     = (bf16_t*)(ws + (size_t)(32u<<20));
    bf16_t* Y2     = (bf16_t*)(ws + (size_t)(48u<<20));
    bf16_t* Wb     = (bf16_t*)(ws + (size_t)(64u<<20));
    bf16_t* Wb_in  = Wb;
    bf16_t* Wb_out = Wb + HH*CC;

    prep_weights_kernel<<<(2*HH*CC)/256, 256, 0, stream>>>(W_in, W_out, Wb);
    adj_softmax_kernel<<<NN, 256, 0, stream>>>(E1, E2, A_bf);
    linear_in_kernel<<<(BB*NN)/128, 256, 0, stream>>>(z, Wb_in, b_in, hT);
    // T2 = 2*A@A - I
    gemm_bf16_kernel<<<dim3(NN/64, NN/64), 256, 0, stream>>>(
        A_bf, A_bf, T2_bf, NN, NN, NN, NN, 1);
    // Y1 = A @ hT
    gemm_bf16_kernel<<<dim3(BH/64, NN/64), 256, 0, stream>>>(
        A_bf, hT, Y1, NN, NN, BH, BH, 0);
    // Y2 = T2 @ hT
    gemm_bf16_kernel<<<dim3(BH/64, NN/64), 256, 0, stream>>>(
        T2_bf, hT, Y2, NN, NN, BH, BH, 0);
    node_kernel<<<NN, 256, 0, stream>>>(hT, Y1, Y2, E1, E2, wp1, wp2,
                                        bp1, bp2, Wb_out, b_out, out);
}
